// StrideMemoryTransformer_80453327389041
// MI455X (gfx1250) — matrix operand order repaired and checked by reading, not compiled, not run on hardware
//
#include <hip/hip_runtime.h>
#include <cstdint>
#include <cstddef>

typedef _Float16 half_t;
typedef __attribute__((ext_vector_type(16))) _Float16 v16h;
typedef __attribute__((ext_vector_type(8)))  _Float16 v8h;
typedef __attribute__((ext_vector_type(8)))  float    v8f;

union V8F { v8f v; float f[8]; };

// ---- model dims (from reference) ----
constexpr int kB = 4, kS = 128, kV = 32000, kD = 768, kH = 12, kL = 2, kF = 3072;
constexpr int kNM = 16, kMI = 48, kStride = 16, kW = kNM + kMI;   // W = 64
constexpr int kDh = kD / kH;                                      // 64
constexpr int kLS = 2, kDI = 1536, kDC = 4, kDS = 16, kDK = 128;

// ---------------- device helpers ----------------
__device__ __forceinline__ float geluf(float x) {
  const float c = 0.7978845608028654f;
  float x3 = x * x * x;
  return 0.5f * x * (1.0f + tanhf(c * (x + 0.044715f * x3)));
}
__device__ __forceinline__ float siluf(float x) { return x / (1.0f + expf(-x)); }
__device__ __forceinline__ float softplusf(float x) {
  return (x > 20.0f) ? x : log1pf(expf(x));
}
__device__ __forceinline__ v16h cat16(v8h lo, v8h hiv) {
  return __builtin_shufflevector(lo, hiv, 0, 1, 2, 3, 4, 5, 6, 7,
                                 8, 9, 10, 11, 12, 13, 14, 15);
}

// ---------------- utility kernels ----------------
__global__ void k_f32_to_f16(const float* __restrict__ src, half_t* __restrict__ dst, size_t n) {
  size_t i = (size_t)blockIdx.x * blockDim.x + threadIdx.x;
  size_t stride = (size_t)gridDim.x * blockDim.x;
  for (; i < n; i += stride) dst[i] = (half_t)src[i];
}
__global__ void k_zero_f32(float* __restrict__ p, size_t n) {
  size_t i = (size_t)blockIdx.x * blockDim.x + threadIdx.x;
  size_t stride = (size_t)gridDim.x * blockDim.x;
  for (; i < n; i += stride) p[i] = 0.0f;
}
__global__ void k_zero_f16(half_t* __restrict__ p, size_t n) {
  size_t i = (size_t)blockIdx.x * blockDim.x + threadIdx.x;
  size_t stride = (size_t)gridDim.x * blockDim.x;
  for (; i < n; i += stride) p[i] = (half_t)0.0f;
}

// transpose one weight slice: src f32 (K x N) row-major -> dst f16 (N x K) row-major
// block (32,8), grid (N/32, K/32); LDS-tiled to keep both sides coalesced.
__global__ void k_transpose_f16(const float* __restrict__ src, half_t* __restrict__ dst,
                                int K, int N) {
  __shared__ float tile[32][33];
  const int tx = threadIdx.x, ty = threadIdx.y;
  const int n0 = blockIdx.x * 32;
  const int k0 = blockIdx.y * 32;
  for (int i = 0; i < 32; i += 8) {
    int k = k0 + ty + i, n = n0 + tx;
    if (k < K && n < N) tile[ty + i][tx] = src[(size_t)k * N + n];
  }
  __syncthreads();
  for (int i = 0; i < 32; i += 8) {
    int n = n0 + ty + i, k = k0 + tx;
    if (n < N && k < K) dst[(size_t)n * K + k] = (half_t)tile[tx][ty + i];
  }
}

// embedding gather: x_emb[b,s,:] = emb[ids[b,s],:]
__global__ void k_embed(const int* __restrict__ ids, const float* __restrict__ emb,
                        float* __restrict__ x_emb, size_t n) {
  size_t i = (size_t)blockIdx.x * blockDim.x + threadIdx.x;
  if (i >= n) return;
  int d  = (int)(i % kD);
  int bs = (int)(i / kD);
  x_emb[i] = emb[(size_t)ids[bs] * kD + d];
}

// shift input buffer, build window and working hidden state
__global__ void k_build(const float* __restrict__ ssm_buf, const float* __restrict__ inp_tmp,
                        const float* __restrict__ x_emb, float* __restrict__ inp_buf,
                        float* __restrict__ win, float* __restrict__ h, int t) {
  size_t i = (size_t)blockIdx.x * blockDim.x + threadIdx.x;
  const size_t n = (size_t)kB * kW * kD;
  if (i >= n) return;
  int d = (int)(i % kD);
  int w = (int)((i / kD) % kW);
  int b = (int)(i / ((size_t)kD * kW));
  float v;
  if (w < kNM) {
    v = ssm_buf[((size_t)b * kNM + w) * kD + d];
  } else {
    int j = w - kNM;
    if (j < kMI - 1) v = inp_tmp[((size_t)b * kMI + j + 1) * kD + d];
    else             v = x_emb[((size_t)b * kS + t) * kD + d];
    inp_buf[((size_t)b * kMI + j) * kD + d] = v;
  }
  win[i] = v;
  h[i]   = v;
}

// LayerNorm over D, f16 output (one 256-thread block per row, rows = B*W)
__global__ void k_layernorm(const float* __restrict__ X, const float* __restrict__ sc,
                            const float* __restrict__ bi, half_t* __restrict__ Y) {
  const int row = blockIdx.x;
  const int tid = threadIdx.x;
  const float* x = X + (size_t)row * kD;
  __shared__ float red[256];
  float s = 0.0f;
  for (int d = tid; d < kD; d += 256) s += x[d];
  red[tid] = s; __syncthreads();
  for (int k = 128; k > 0; k >>= 1) { if (tid < k) red[tid] += red[tid + k]; __syncthreads(); }
  float mu = red[0] / kD; __syncthreads();
  float v = 0.0f;
  for (int d = tid; d < kD; d += 256) { float t = x[d] - mu; v += t * t; }
  red[tid] = v; __syncthreads();
  for (int k = 128; k > 0; k >>= 1) { if (tid < k) red[tid] += red[tid + k]; __syncthreads(); }
  float inv = rsqrtf(red[0] / kD + 1e-5f);
  for (int d = tid; d < kD; d += 256)
    Y[(size_t)row * kD + d] = (half_t)((x[d] - mu) * inv * sc[d] + bi[d]);
}

// final LN of last window row only -> lm head input (rows 0..3 of a 16xD f16 tile)
__global__ void k_lnf_last(const float* __restrict__ hbuf, const float* __restrict__ sc,
                           const float* __restrict__ bi, half_t* __restrict__ lm_in) {
  const int b = blockIdx.x;
  const int tid = threadIdx.x;
  const float* x = hbuf + ((size_t)b * kW + (kW - 1)) * kD;
  __shared__ float red[256];
  float s = 0.0f;
  for (int d = tid; d < kD; d += 256) s += x[d];
  red[tid] = s; __syncthreads();
  for (int k = 128; k > 0; k >>= 1) { if (tid < k) red[tid] += red[tid + k]; __syncthreads(); }
  float mu = red[0] / kD; __syncthreads();
  float v = 0.0f;
  for (int d = tid; d < kD; d += 256) { float t = x[d] - mu; v += t * t; }
  red[tid] = v; __syncthreads();
  for (int k = 128; k > 0; k >>= 1) { if (tid < k) red[tid] += red[tid + k]; __syncthreads(); }
  float inv = rsqrtf(red[0] / kD + 1e-5f);
  for (int d = tid; d < kD; d += 256)
    lm_in[(size_t)b * kD + d] = (half_t)((x[d] - mu) * inv * sc[d] + bi[d]);
}

// ---------------- WMMA GEMM: C(MxN) = act(A(MxK) @ B(KxN) + bias [+ Cres]) ----------------
// A f16 row-major (MxK); Bt f16 is B TRANSPOSED (NxK row-major) so each lane's
// B-fragment (one column, K in two runs of 8) is two contiguous b128 loads.
// One wave per 16x16 C tile, 4 waves/block along N.
__global__ void k_gemm_f16(const half_t* __restrict__ A, const half_t* __restrict__ Bt,
                           const float* __restrict__ bias,
                           float* __restrict__ C, const float* __restrict__ Cres,
                           half_t* __restrict__ Ch,
                           int M, int N, int K, int act) {
  const int lane  = threadIdx.x & 31;
  const int wave  = threadIdx.x >> 5;
  const int tileN = blockIdx.x * 4 + wave;
  const int tileM = blockIdx.y;
  if (tileN * 16 >= N) return;           // wave-uniform: EXEC stays all-ones for WMMA
  const int r16 = lane & 15;
  const int hi  = lane >> 4;
  const half_t* Arow = A  + (size_t)(tileM * 16 + r16) * K;
  const int col = tileN * 16 + r16;
  const half_t* Brow = Bt + (size_t)col * K;
  V8F acc; acc.v = (v8f){0.f, 0.f, 0.f, 0.f, 0.f, 0.f, 0.f, 0.f};
#pragma unroll 4
  for (int k0 = 0; k0 < K; k0 += 32) {
    // A fragment: lanes 0-15 -> K {0..7,16..23}, lanes 16-31 -> K {8..15,24..31}
    v8h a0 = *(const v8h*)(Arow + k0 + hi * 8);
    v8h a1 = *(const v8h*)(Arow + k0 + 16 + hi * 8);
    // B fragment: lane holds column `col`, K = k0 + hi*8 + [0..7], k0 + 16 + hi*8 + [0..7] (same as A)
    const int kb = k0 + hi * 8;
    v8h b0 = *(const v8h*)(Brow + kb);
    v8h b1 = *(const v8h*)(Brow + kb + 16);
    if (k0 + 128 < K) {
      // locality 3 -> WGP-scope prefetch: pulls into all cache levels (weights are
      // already L2-resident; we want them staged into the WGP cache ahead of use).
      __builtin_prefetch(Arow + k0 + 128, 0, 3);
      __builtin_prefetch(Brow + k0 + 128, 0, 3);
    }
    acc.v = __builtin_amdgcn_wmma_f32_16x16x32_f16(false, cat16(a0, a1), false,
                                                   cat16(b0, b1), (short)0, acc.v,
                                                   false, false);
  }
  const int mBase = tileM * 16 + hi * 8;
  const float bval = bias ? bias[col] : 0.0f;
#pragma unroll
  for (int r = 0; r < 8; ++r) {
    int m = mBase + r;
    size_t idx = (size_t)m * N + col;
    float v = acc.f[r] + bval;
    if (Cres) v += Cres[idx];
    if (act == 1) v = geluf(v);
    if (C)  C[idx]  = v;
    if (Ch) Ch[idx] = (half_t)v;
  }
}

// ---------------- WMMA lm_head: logits(B,V) = lm_in(16xD) @ emb^T ----------------
// Tied embedding row-major (VxD) is already the NxK layout => contiguous B loads.
__global__ void k_lmhead(const half_t* __restrict__ Ain, const half_t* __restrict__ Eh,
                         float* __restrict__ out, int t) {
  const int lane  = threadIdx.x & 31;
  const int wave  = threadIdx.x >> 5;
  const int tileN = blockIdx.x * 4 + wave;
  if (tileN * 16 >= kV) return;
  const int r16 = lane & 15;
  const int hi  = lane >> 4;
  const half_t* Arow = Ain + (size_t)r16 * kD;
  const int col = tileN * 16 + r16;
  const half_t* Brow = Eh + (size_t)col * kD;
  V8F acc; acc.v = (v8f){0.f, 0.f, 0.f, 0.f, 0.f, 0.f, 0.f, 0.f};
#pragma unroll 4
  for (int k0 = 0; k0 < kD; k0 += 32) {
    v8h a0 = *(const v8h*)(Arow + k0 + hi * 8);
    v8h a1 = *(const v8h*)(Arow + k0 + 16 + hi * 8);
    const int kb = k0 + hi * 8;
    v8h b0 = *(const v8h*)(Brow + kb);
    v8h b1 = *(const v8h*)(Brow + kb + 16);
    if (k0 + 128 < kD) __builtin_prefetch(Brow + k0 + 128, 0, 3);
    acc.v = __builtin_amdgcn_wmma_f32_16x16x32_f16(false, cat16(a0, a1), false,
                                                   cat16(b0, b1), (short)0, acc.v,
                                                   false, false);
  }
  if (hi == 0) {             // lanes 0-15 hold rows 0..7; rows 0..3 are batch b
#pragma unroll
    for (int r = 0; r < kB; ++r)
      out[(size_t)r * kS * kV + (size_t)t * kV + col] = acc.f[r];
  }
}

// ---------------- attention (per (b,h) head; W=64 threads) ----------------
__global__ void k_attn(const float* __restrict__ qkv, half_t* __restrict__ o_h) {
  const int b = blockIdx.x / kH;
  const int h = blockIdx.x % kH;
  const int tid = threadIdx.x;  // 0..63 : row index
  __shared__ float sQ[kW][kDh + 1], sK[kW][kDh + 1], sV[kW][kDh + 1];
  const float* base = qkv + (size_t)b * kW * 3 * kD;
  for (int d = 0; d < kDh; ++d) {
    sQ[tid][d] = base[(size_t)tid * 3 * kD + h * kDh + d];
    sK[tid][d] = base[(size_t)tid * 3 * kD + kD + h * kDh + d];
    sV[tid][d] = base[(size_t)tid * 3 * kD + 2 * kD + h * kDh + d];
  }
  __syncthreads();
  float scr[kW];
  float mx = -1e30f;
  for (int j = 0; j <= tid; ++j) {
    float s = 0.0f;
    for (int d = 0; d < kDh; ++d) s += sQ[tid][d] * sK[j][d];
    s *= 0.125f;  // 1/sqrt(64)
    scr[j] = s;
    mx = fmaxf(mx, s);
  }
  float sum = 0.0f;
  for (int j = 0; j <= tid; ++j) { float e = expf(scr[j] - mx); scr[j] = e; sum += e; }
  float inv = 1.0f / sum;
  half_t* op = o_h + ((size_t)b * kW + tid) * kD + h * kDh;
  for (int d = 0; d < kDh; ++d) {
    float a = 0.0f;
    for (int j = 0; j <= tid; ++j) a += scr[j] * sV[j][d];
    op[d] = (half_t)(a * inv);
  }
}

// ---------------- pooling ----------------
__global__ void k_pool_prep(const float* __restrict__ wk, const float* __restrict__ q,
                            float* __restrict__ u) {
  int d = blockIdx.x * blockDim.x + threadIdx.x;
  if (d >= kD) return;
  float s = 0.0f;
  for (int j = 0; j < kDK; ++j) s += wk[(size_t)d * kDK + j] * q[j];
  u[d] = s * rsqrtf((float)kDK);
}
__global__ void k_pool(const float* __restrict__ win, const float* __restrict__ u,
                       float* __restrict__ pooled) {
  const int b = blockIdx.x;
  const int w = threadIdx.x;  // 0..63
  __shared__ float sc[kW];
  const float* xw = win + ((size_t)b * kW + w) * kD;
  float s = 0.0f;
  for (int d = 0; d < kD; ++d) s += xw[d] * u[d];
  sc[w] = s; __syncthreads();
  float m = -1e30f;
  for (int j = 0; j < kW; ++j) m = fmaxf(m, sc[j]);
  float e = expf(s - m);
  __syncthreads();
  sc[w] = e; __syncthreads();
  float sum = 0.0f;
  for (int j = 0; j < kW; ++j) sum += sc[j];
  float invs = 1.0f / sum;
  for (int d = w; d < kD; d += kW) {
    float a = 0.0f;
    for (int j = 0; j < kW; ++j) a += sc[j] * win[((size_t)b * kW + j) * kD + d];
    pooled[(size_t)b * kD + d] = a * invs;
  }
}

// ---------------- Mamba-style SSM step (one block per batch element) ----------------
__global__ void k_ssm(const float* __restrict__ pooled, float* __restrict__ conv_c,
                      float* __restrict__ h_st, float* __restrict__ s_out,
                      const float* __restrict__ rms, const float* __restrict__ winw,
                      const float* __restrict__ convw, const float* __restrict__ convb,
                      const float* __restrict__ wdt, const float* __restrict__ bdt,
                      const float* __restrict__ wb, const float* __restrict__ wc,
                      const float* __restrict__ alog, const float* __restrict__ dskip,
                      const float* __restrict__ wout, int do_write) {
  const int b = blockIdx.x;
  const int tid = threadIdx.x;  // 256 threads
  __shared__ float x[kD], resid[kD];
  __shared__ float xi[kDI], zz[kDI], xc[kDI], yv[kDI];
  __shared__ float red[256];
  __shared__ float Bm[kDS], Cm[kDS];
  __shared__ float invn;
  for (int d = tid; d < kD; d += 256) x[d] = pooled[(size_t)b * kD + d];
  __syncthreads();
  for (int l = 0; l < kLS; ++l) {
    float ss = 0.0f;
    for (int d = tid; d < kD; d += 256) { resid[d] = x[d]; ss += x[d] * x[d]; }
    red[tid] = ss; __syncthreads();
    for (int k = 128; k > 0; k >>= 1) { if (tid < k) red[tid] += red[tid + k]; __syncthreads(); }
    if (tid == 0) invn = rsqrtf(red[0] / kD + 1e-6f);
    __syncthreads();
    float inv = invn;
    for (int d = tid; d < kD; d += 256) x[d] = x[d] * inv * rms[(size_t)l * kD + d];
    __syncthreads();
    const float* Wl = winw + (size_t)l * kD * 2 * kDI;
    for (int i = tid; i < 2 * kDI; i += 256) {
      float s = 0.0f;
      for (int d = 0; d < kD; ++d) s += x[d] * Wl[(size_t)d * 2 * kDI + i];
      if (i < kDI) xi[i] = s; else zz[i - kDI] = s;
    }
    __syncthreads();
    if (tid < kDS) { Bm[tid] = 0.0f; Cm[tid] = 0.0f; }
    __syncthreads();
    float pB[kDS], pC[kDS];
#pragma unroll
    for (int s2 = 0; s2 < kDS; ++s2) { pB[s2] = 0.0f; pC[s2] = 0.0f; }
    for (int i = tid; i < kDI; i += 256) {
      float* cc = conv_c + (((size_t)l * kB + b) * kDI + i) * kDC;
      float c0 = cc[1], c1 = cc[2], c2 = cc[3], c3 = xi[i];
      const float* cw = convw + ((size_t)l * kDI + i) * kDC;
      float pre = c0 * cw[0] + c1 * cw[1] + c2 * cw[2] + c3 * cw[3] + convb[(size_t)l * kDI + i];
      float v = siluf(pre);
      xc[i] = v;
      if (do_write) { cc[0] = c0; cc[1] = c1; cc[2] = c2; cc[3] = c3; }
      const float* wbp = wb + ((size_t)l * kDI + i) * kDS;
      const float* wcp = wc + ((size_t)l * kDI + i) * kDS;
#pragma unroll
      for (int s2 = 0; s2 < kDS; ++s2) { pB[s2] += v * wbp[s2]; pC[s2] += v * wcp[s2]; }
    }
#pragma unroll
    for (int s2 = 0; s2 < kDS; ++s2) { atomicAdd(&Bm[s2], pB[s2]); atomicAdd(&Cm[s2], pC[s2]); }
    __syncthreads();
    for (int i = tid; i < kDI; i += 256) {
      float v = xc[i];
      float dtv = softplusf(v * wdt[(size_t)l * kDI + i] + bdt[(size_t)l * kDI + i]);
      float* hs = h_st + (((size_t)l * kB + b) * kDI + i) * kDS;
      const float* al = alog + ((size_t)l * kDI + i) * kDS;
      float y = 0.0f;
#pragma unroll
      for (int s2 = 0; s2 < kDS; ++s2) {
        float A = -expf(al[s2]);
        float hn = expf(dtv * A) * hs[s2] + dtv * Bm[s2] * v;
        if (do_write) hs[s2] = hn;
        y += hn * Cm[s2];
      }
      y += dskip[(size_t)l * kDI + i] * v;
      yv[i] = y * siluf(zz[i]);
    }
    __syncthreads();
    const float* Wo = wout + (size_t)l * kDI * kD;
    for (int d = tid; d < kD; d += 256) {
      float s = 0.0f;
      for (int i = 0; i < kDI; ++i) s += yv[i] * Wo[(size_t)i * kD + d];
      x[d] = resid[d] + s;
    }
    __syncthreads();
  }
  for (int d = tid; d < kD; d += 256) s_out[(size_t)b * kD + d] = x[d];
}

// shift ssm memory buffer (only on write steps): new[j] = old[j+1], new[last] = s_out
__global__ void k_shift_ssm(const float* __restrict__ tmp, const float* __restrict__ s_out,
                            float* __restrict__ ssm_buf) {
  size_t i = (size_t)blockIdx.x * blockDim.x + threadIdx.x;
  const size_t n = (size_t)kB * kNM * kD;
  if (i >= n) return;
  int d = (int)(i % kD);
  int j = (int)((i / kD) % kNM);
  int b = (int)(i / ((size_t)kD * kNM));
  float v = (j < kNM - 1) ? tmp[((size_t)b * kNM + j + 1) * kD + d]
                          : s_out[(size_t)b * kD + d];
  ssm_buf[i] = v;
}

// ---------------- host side ----------------
static inline size_t alignup(size_t x) { return (x + 255) & ~(size_t)255; }

extern "C" void kernel_launch(void* const* d_in, const int* in_sizes, int n_in,
                              void* d_out, int out_size, void* d_ws, size_t ws_size,
                              hipStream_t stream) {
  (void)in_sizes; (void)n_in; (void)out_size; (void)ws_size;
  const int*   ids     = (const int*)  d_in[0];
  const float* emb     = (const float*)d_in[1];
  const float* ln1_s   = (const float*)d_in[2];
  const float* ln1_b   = (const float*)d_in[3];
  const float* wqkv    = (const float*)d_in[4];
  const float* bqkv    = (const float*)d_in[5];
  const float* wo      = (const float*)d_in[6];
  const float* bo      = (const float*)d_in[7];
  const float* ln2_s   = (const float*)d_in[8];
  const float* ln2_b   = (const float*)d_in[9];
  const float* w1      = (const float*)d_in[10];
  const float* b1      = (const float*)d_in[11];
  const float* w2      = (const float*)d_in[12];
  const float* b2      = (const float*)d_in[13];
  const float* lnf_s   = (const float*)d_in[14];
  const float* lnf_b   = (const float*)d_in[15];
  const float* pool_wk = (const float*)d_in[16];
  const float* pool_q  = (const float*)d_in[17];
  const float* ssm_rms = (const float*)d_in[18];
  const float* ssm_win = (const float*)d_in[19];
  const float* ssm_cw  = (const float*)d_in[20];
  const float* ssm_cb  = (const float*)d_in[21];
  const float* ssm_wdt = (const float*)d_in[22];
  const float* ssm_bdt = (const float*)d_in[23];
  const float* ssm_wb  = (const float*)d_in[24];
  const float* ssm_wc  = (const float*)d_in[25];
  const float* ssm_al  = (const float*)d_in[26];
  const float* ssm_dsk = (const float*)d_in[27];
  const float* ssm_wou = (const float*)d_in[28];
  float* out = (float*)d_out;

  // workspace arena
  char* w = (char*)d_ws;
  size_t off = 0;
  auto alloc = [&](size_t bytes) { void* p = w + off; off = alignup(off + bytes); return p; };

  half_t* emb_h   = (half_t*)alloc((size_t)kV * kD * 2);          // V x D (already NxK)
  half_t* wqkvT   = (half_t*)alloc((size_t)kL * 3 * kD * kD * 2); // per l: 3D x D
  half_t* woT     = (half_t*)alloc((size_t)kL * kD * kD * 2);     // per l: D x D
  half_t* w1T     = (half_t*)alloc((size_t)kL * kF * kD * 2);     // per l: F x D
  half_t* w2T     = (half_t*)alloc((size_t)kL * kD * kF * 2);     // per l: D x F
  float*  x_emb   = (float*)alloc((size_t)kB * kS * kD * 4);
  float*  inp_buf = (float*)alloc((size_t)kB * kMI * kD * 4);
  float*  inp_tmp = (float*)alloc((size_t)kB * kMI * kD * 4);
  float*  ssm_buf = (float*)alloc((size_t)kB * kNM * kD * 4);
  float*  ssm_tmp = (float*)alloc((size_t)kB * kNM * kD * 4);
  float*  win     = (float*)alloc((size_t)kB * kW * kD * 4);
  float*  hbuf    = (float*)alloc((size_t)kB * kW * kD * 4);
  half_t* y_h     = (half_t*)alloc((size_t)kB * kW * kD * 2);
  float*  qkv     = (float*)alloc((size_t)kB * kW * 3 * kD * 4);
  half_t* o_h     = (half_t*)alloc((size_t)kB * kW * kD * 2);
  half_t* mid_h   = (half_t*)alloc((size_t)kB * kW * kF * 2);
  half_t* lm_in   = (half_t*)alloc((size_t)16 * kD * 2);
  float*  u_vec   = (float*)alloc((size_t)kD * 4);
  float*  pooled  = (float*)alloc((size_t)kB * kD * 4);
  float*  s_out   = (float*)alloc((size_t)kB * kD * 4);
  float*  conv_c  = (float*)alloc((size_t)kLS * kB * kDI * kDC * 4);
  float*  h_st    = (float*)alloc((size_t)kLS * kB * kDI * kDS * 4);

  // ---- one-time prep: f16 conversion + weight transposes (constant across scan) ----
  k_f32_to_f16<<<4096, 256, 0, stream>>>(emb, emb_h, (size_t)kV * kD);
  dim3 tb(32, 8);
  for (int l = 0; l < kL; ++l) {
    k_transpose_f16<<<dim3(3 * kD / 32, kD / 32), tb, 0, stream>>>(
        wqkv + (size_t)l * kD * 3 * kD, wqkvT + (size_t)l * 3 * kD * kD, kD, 3 * kD);
    k_transpose_f16<<<dim3(kD / 32, kD / 32), tb, 0, stream>>>(
        wo + (size_t)l * kD * kD, woT + (size_t)l * kD * kD, kD, kD);
    k_transpose_f16<<<dim3(kF / 32, kD / 32), tb, 0, stream>>>(
        w1 + (size_t)l * kD * kF, w1T + (size_t)l * kF * kD, kD, kF);
    k_transpose_f16<<<dim3(kD / 32, kF / 32), tb, 0, stream>>>(
        w2 + (size_t)l * kF * kD, w2T + (size_t)l * kD * kF, kF, kD);
  }
  {
    size_t n = (size_t)kB * kS * kD;
    k_embed<<<(unsigned)((n + 255) / 256), 256, 0, stream>>>(ids, emb, x_emb, n);
  }
  k_zero_f32<<<256, 256, 0, stream>>>(inp_buf, (size_t)kB * kMI * kD);
  k_zero_f32<<<256, 256, 0, stream>>>(ssm_buf, (size_t)kB * kNM * kD);
  k_zero_f32<<<256, 256, 0, stream>>>(conv_c,  (size_t)kLS * kB * kDI * kDC);
  k_zero_f32<<<256, 256, 0, stream>>>(h_st,    (size_t)kLS * kB * kDI * kDS);
  k_zero_f16<<<64,  256, 0, stream>>>(lm_in,   (size_t)16 * kD);
  k_pool_prep<<<(kD + 255) / 256, 256, 0, stream>>>(pool_wk, pool_q, u_vec);

  auto gemm = [&](const half_t* A, const half_t* Bt, const float* bias, float* C,
                  const float* Cres, half_t* Ch, int M, int N, int K, int act) {
    dim3 g((unsigned)((N / 16 + 3) / 4), (unsigned)(M / 16));
    k_gemm_f16<<<g, 128, 0, stream>>>(A, Bt, bias, C, Cres, Ch, M, N, K, act);
  };

  const int M = kB * kW;  // 256 rows
  // ---- sequential scan ----
  for (int t = 0; t < kS; ++t) {
    hipMemcpyAsync(inp_tmp, inp_buf, (size_t)kB * kMI * kD * 4,
                   hipMemcpyDeviceToDevice, stream);
    {
      size_t n = (size_t)kB * kW * kD;
      k_build<<<(unsigned)((n + 255) / 256), 256, 0, stream>>>(
          ssm_buf, inp_tmp, x_emb, inp_buf, win, hbuf, t);
    }
    for (int l = 0; l < kL; ++l) {
      k_layernorm<<<M, 256, 0, stream>>>(hbuf, ln1_s + (size_t)l * kD, ln1_b + (size_t)l * kD, y_h);
      gemm(y_h, wqkvT + (size_t)l * 3 * kD * kD, bqkv + (size_t)l * 3 * kD,
           qkv, nullptr, nullptr, M, 3 * kD, kD, 0);
      k_attn<<<kB * kH, kW, 0, stream>>>(qkv, o_h);
      gemm(o_h, woT + (size_t)l * kD * kD, bo + (size_t)l * kD,
           hbuf, hbuf, nullptr, M, kD, kD, 0);
      k_layernorm<<<M, 256, 0, stream>>>(hbuf, ln2_s + (size_t)l * kD, ln2_b + (size_t)l * kD, y_h);
      gemm(y_h, w1T + (size_t)l * kF * kD, b1 + (size_t)l * kF,
           nullptr, nullptr, mid_h, M, kF, kD, 1);
      gemm(mid_h, w2T + (size_t)l * kD * kF, b2 + (size_t)l * kD,
           hbuf, hbuf, nullptr, M, kD, kF, 0);
    }
    k_lnf_last<<<kB, 256, 0, stream>>>(hbuf, lnf_s, lnf_b, lm_in);
    k_lmhead<<<kV / 16 / 4, 128, 0, stream>>>(lm_in, emb_h, out, t);

    const bool wr = (t > 0) && (t % kStride == 0);
    if (wr) {
      k_pool<<<kB, kW, 0, stream>>>(win, u_vec, pooled);
      k_ssm<<<kB, 256, 0, stream>>>(pooled, conv_c, h_st, s_out, ssm_rms, ssm_win,
                                    ssm_cw, ssm_cb, ssm_wdt, ssm_bdt, ssm_wb, ssm_wc,
                                    ssm_al, ssm_dsk, ssm_wou, 1);
      hipMemcpyAsync(ssm_tmp, ssm_buf, (size_t)kB * kNM * kD * 4,
                     hipMemcpyDeviceToDevice, stream);
      size_t n = (size_t)kB * kNM * kD;
      k_shift_ssm<<<(unsigned)((n + 255) / 256), 256, 0, stream>>>(ssm_tmp, s_out, ssm_buf);
    }
  }
}
